// VectorQuantizer_6597069767085
// MI455X (gfx1250) — matrix operand order repaired and checked by reading, not compiled, not run on hardware
//
#include <hip/hip_runtime.h>
#include <cstdint>
#include <cstddef>

// ---------------- problem constants ----------------
#define N_TOK   16384            // 16*1024 rows of z
#define E_DIM   256              // feature dim (K)
#define N_E     8192             // codebook size (N)
#define Z_ELEMS (N_TOK * E_DIM)  // 4194304
#define E_ELEMS (N_E * E_DIM)    // 2097152

// ---------------- d_out layout (float elements, tuple concat order) ------
// loss(1) | z_q_st(4194304) | perplexity(1) | min_encodings(134217728) | indices(16384)
#define OUT_LOSS 0ull
#define OUT_ZQ   1ull
#define OUT_PERP 4194305ull
#define OUT_ME   4194306ull
#define OUT_IDX  138412034ull
#define ME_ELEMS 134217728ull

// Scratch lives INSIDE the min_encodings region (537MB) and is wiped by the
// zero-fill pass that runs after the GEMM. 256B-aligned byte offset.
#define SCR_BASE 16777472ull

typedef __bf16 bf16;
typedef __attribute__((ext_vector_type(16))) bf16  v16bf;
typedef __attribute__((ext_vector_type(8)))  bf16  v8bf;
typedef __attribute__((ext_vector_type(8)))  float v8f;

// ---------------- kernel 1: f32 -> split bf16 (hi + lo) ----------------
__global__ void vq_convert(const float* __restrict__ src,
                           bf16* __restrict__ hi, bf16* __restrict__ lo, int n) {
    int i = blockIdx.x * blockDim.x + threadIdx.x;
    if (i < n) {
        float x = src[i];
        bf16 h = (bf16)x;                 // RNE
        hi[i] = h;
        lo[i] = (bf16)(x - (float)h);     // residual
    }
}

// ---------------- kernel 2: exact f32 codebook row norms ||e_n||^2 -------
__global__ void vq_rownorm(const float* __restrict__ e, float* __restrict__ e2) {
    int wid  = (blockIdx.x * blockDim.x + threadIdx.x) >> 5;   // one wave per row
    int lane = threadIdx.x & 31;
    const float4* p = (const float4*)(e + (size_t)wid * E_DIM);
    float4 a = p[2 * lane], b = p[2 * lane + 1];
    float s = a.x*a.x + a.y*a.y + a.z*a.z + a.w*a.w
            + b.x*b.x + b.y*b.y + b.z*b.z + b.w*b.w;
    #pragma unroll
    for (int m = 1; m < 32; m <<= 1) s += __shfl_xor(s, m, 32);
    if (lane == 0) e2[wid] = s;
}

// ---------------- kernel 3: fused split-bf16 WMMA GEMM + argmin ----------
// Each wave owns a 16-row M tile (A hi/lo resident in 128 VGPRs for all of
// K=256) and loops over all 512 N-tiles of the codebook. Per tile:
// 8 K-chunks x 3 chained v_wmma_f32_16x16x32_bf16 (hi*hi + hi*lo + lo*hi).
// Epilogue tracks per-lane running min of ||e||^2 - 2*dot, reduced across the
// 16-lane N groups at the end with xor-shuffles (first-index tie-break).
__global__ void __launch_bounds__(256)
vq_gemm_argmin(const bf16* __restrict__ zhi, const bf16* __restrict__ zlo,
               const bf16* __restrict__ ehi, const bf16* __restrict__ elo,
               const float* __restrict__ e2, float* __restrict__ out_idx) {
    const int tid  = threadIdx.x;
    const int lane = tid & 31;
    const int h    = lane >> 4;           // half-wave (K-phase / M-phase)
    const int lr   = lane & 15;           // N within tile / M row selector
    const int wave = blockIdx.x * (blockDim.x >> 5) + (tid >> 5);
    const int m0   = wave * 16;

    // --- resident A fragments: contiguous k order, k = 16h + s (same as B) ---
    // lanes 0-15 : VGPR0-7 -> K 0..15
    // lanes 16-31: VGPR0-7 -> K 16..31
    v16bf Ah[8], Al[8];
    const bf16* zr_hi = zhi + (size_t)(m0 + lr) * E_DIM;
    const bf16* zr_lo = zlo + (size_t)(m0 + lr) * E_DIM;
    #pragma unroll
    for (int kc = 0; kc < 8; ++kc) {
        const int kb = kc * 32 + h * 16;
        union { v16bf v; v8bf p[2]; } uh, ul;
        uh.p[0] = *(const v8bf*)(zr_hi + kb);
        uh.p[1] = *(const v8bf*)(zr_hi + kb + 8);
        ul.p[0] = *(const v8bf*)(zr_lo + kb);
        ul.p[1] = *(const v8bf*)(zr_lo + kb + 8);
        Ah[kc] = uh.v;
        Al[kc] = ul.v;
    }

    float minv[8];
    int   mini[8];
    #pragma unroll
    for (int v = 0; v < 8; ++v) { minv[v] = 3.0e38f; mini[v] = 0; }

    for (int nt = 0; nt < N_E / 16; ++nt) {
        const int n = nt * 16 + lr;       // this lane's codebook entry
        // B in the same contiguous k order as A: lanes 0-15 K 0..15, lanes 16-31 K 16..31,
        // column (N) = lane%16 -> one contiguous 32B load per lane per chunk.
        const bf16* er_hi = ehi + (size_t)n * E_DIM + h * 16;
        const bf16* er_lo = elo + (size_t)n * E_DIM + h * 16;
        v8f C = {0.f, 0.f, 0.f, 0.f, 0.f, 0.f, 0.f, 0.f};
        #pragma unroll
        for (int kc = 0; kc < 8; ++kc) {
            v16bf bh = *(const v16bf*)(er_hi + kc * 32);
            v16bf bl = *(const v16bf*)(er_lo + kc * 32);
            C = __builtin_amdgcn_wmma_f32_16x16x32_bf16(false, Ah[kc], false, bh,
                                                        (short)0, C, false, false);
            C = __builtin_amdgcn_wmma_f32_16x16x32_bf16(false, Ah[kc], false, bl,
                                                        (short)0, C, false, false);
            C = __builtin_amdgcn_wmma_f32_16x16x32_bf16(false, Al[kc], false, bh,
                                                        (short)0, C, false, false);
        }
        // score = ||e_n||^2 - 2 * <z_m, e_n>   (||z_m||^2 is argmin-invariant)
        const float en2 = e2[n];
        #pragma unroll
        for (int v = 0; v < 8; ++v) {
            float s = fmaf(-2.0f, C[v], en2);
            if (s < minv[v]) { minv[v] = s; mini[v] = n; }
        }
    }

    // cross-lane argmin over the 16 N-lanes of each half-wave.
    // C layout: VGPR v -> M=v (lanes 0-15), M=v+8 (lanes 16-31).
    #pragma unroll
    for (int v = 0; v < 8; ++v) {
        float val = minv[v];
        int   idx = mini[v];
        #pragma unroll
        for (int m = 1; m < 16; m <<= 1) {
            float ov = __shfl_xor(val, m, 32);
            int   oi = __shfl_xor(idx, m, 32);
            if (ov < val || (ov == val && oi < idx)) { val = ov; idx = oi; }
        }
        if (lr == 0) out_idx[m0 + v + 8 * h] = (float)idx;
    }
}

// ---------------- kernel 4: zero min_encodings region (wipes scratch) ----
__global__ void vq_zero_me(float* __restrict__ dout) {
    size_t tid = (size_t)blockIdx.x * blockDim.x + threadIdx.x;
    float2* base = (float2*)(dout + OUT_ME);            // 8B aligned
    const size_t n2 = ME_ELEMS / 2;
    const size_t stride = (size_t)gridDim.x * blockDim.x;
    for (size_t i = tid; i < n2; i += stride) base[i] = make_float2(0.f, 0.f);
    if (tid == 0) dout[OUT_LOSS] = 0.f;
}

// ---------------- kernel 5: gather z_q, loss partials, one-hot scatter ---
__global__ void __launch_bounds__(256)
vq_outputs(const float* __restrict__ z, const float* __restrict__ emb,
           float* __restrict__ dout) {
    const int m = blockIdx.x;
    const int t = threadIdx.x;
    const int idx = (int)dout[OUT_IDX + m];
    const float q = emb[(size_t)idx * E_DIM + t];
    const float x = z[(size_t)m * E_DIM + t];
    dout[OUT_ZQ + (size_t)m * E_DIM + t] = q;   // z_q_st == z_q numerically
    float d = q - x;
    float s = d * d;
    #pragma unroll
    for (int mk = 1; mk < 32; mk <<= 1) s += __shfl_xor(s, mk, 32);
    __shared__ float wsum[8];
    if ((t & 31) == 0) wsum[t >> 5] = s;
    __syncthreads();
    if (t == 0) {
        float tot = 0.f;
        #pragma unroll
        for (int j = 0; j < 8; ++j) tot += wsum[j];
        atomicAdd(&dout[OUT_LOSS], tot);
        dout[OUT_ME + (size_t)m * N_E + idx] = 1.0f;   // one-hot
    }
}

// ---------------- kernel 6: counts histogram in LDS -> perplexity, loss --
__global__ void vq_finalize(float* __restrict__ dout) {
    __shared__ float cnt[N_E];                 // 32KB of the 320KB WGP LDS
    const int t = threadIdx.x;                 // 1024 threads
    for (int b = t; b < N_E; b += 1024) cnt[b] = 0.f;
    __syncthreads();
    for (int i = t; i < N_TOK; i += 1024) {
        int idx = (int)dout[OUT_IDX + i];
        atomicAdd(&cnt[idx], 1.0f);
    }
    __syncthreads();
    float s = 0.f;
    for (int b = t; b < N_E; b += 1024) {
        float p = cnt[b] * (1.0f / N_TOK);
        s -= p * logf(p + 1e-10f);
    }
    #pragma unroll
    for (int mk = 1; mk < 32; mk <<= 1) s += __shfl_xor(s, mk, 32);
    __shared__ float ps[32];
    if ((t & 31) == 0) ps[t >> 5] = s;
    __syncthreads();
    if (t == 0) {
        float tot = 0.f;
        #pragma unroll
        for (int j = 0; j < 32; ++j) tot += ps[j];
        dout[OUT_PERP] = expf(tot);
        dout[OUT_LOSS] = dout[OUT_LOSS] * (1.25f / (float)Z_ELEMS); // (1+beta)*MSE
    }
}

// ---------------- launch -------------------------------------------------
extern "C" void kernel_launch(void* const* d_in, const int* in_sizes, int n_in,
                              void* d_out, int out_size, void* d_ws, size_t ws_size,
                              hipStream_t stream) {
    const float* z   = (const float*)d_in[0];   // [16,1024,256] f32
    const float* emb = (const float*)d_in[1];   // [8192,256]    f32
    float* out = (float*)d_out;

    // scratch carved from the min_encodings output region (zeroed later)
    char* sb = (char*)d_out + SCR_BASE;
    bf16* zhi = (bf16*)sb;
    bf16* zlo = zhi + Z_ELEMS;
    bf16* ehi = zlo + Z_ELEMS;
    bf16* elo = ehi + E_ELEMS;
    float* e2 = (float*)(elo + E_ELEMS);

    vq_convert<<<Z_ELEMS / 256, 256, 0, stream>>>(z, zhi, zlo, Z_ELEMS);
    vq_convert<<<E_ELEMS / 256, 256, 0, stream>>>(emb, ehi, elo, E_ELEMS);
    vq_rownorm<<<N_E / 8, 256, 0, stream>>>(emb, e2);
    vq_gemm_argmin<<<128, 256, 0, stream>>>(zhi, zlo, ehi, elo, e2, out + OUT_IDX);
    vq_zero_me<<<32768, 256, 0, stream>>>(out);
    vq_outputs<<<N_TOK, 256, 0, stream>>>(z, emb, out);
    vq_finalize<<<1, 1024, 0, stream>>>(out);
}
